// ParallelmomResidualLayer_90941637525893
// MI455X (gfx1250) — matrix operand order repaired and checked by reading, not compiled, not run on hardware
//
#include <hip/hip_runtime.h>
#include <cstdint>
#include <cstddef>

// ---------------------------------------------------------------------------
// MI455X (gfx1250, wave32) implementation.
// - All GEMMs + attention on v_wmma_f32_16x16x32_bf16 (f32 accumulate).
// - GEMM B operands staged global->LDS with CDNA5 async-to-LDS instructions
//   (ASYNCcnt pipeline, double buffered), shared by 8 waves per workgroup;
//   each wave register-blocks 2 row tiles (8 wmma per K-step per wave).
// - Attention: one wave per 32 query rows, K/V fragments reused across both
//   row tiles (16 wmma per 32-col chunk), fused per-head LayerNorm.
// - __launch_bounds__ lifts the VGPR cap so fragments stay in registers
//   (round 3 showed scratch spills of the Q fragments without it).
// - global_prefetch_b8 via __builtin_prefetch on streamed operands.
// ---------------------------------------------------------------------------

#define L_SEQ   2048
#define BATCH   2
#define NTOK    (L_SEQ*BATCH)   // 4096 tokens
#define DMODEL  1024
#define HEADS   16
#define DK_DIM  512
#define DV_DIM  1024
#define HDKD    32
#define HDVD    64
#define FF_DIM  2816
#define LN_EPS  1e-5f

typedef __attribute__((ext_vector_type(16))) __bf16 v16bf;
typedef __attribute__((ext_vector_type(8)))  float  v8f;

__device__ __forceinline__ v8f wmma_bf16(v16bf a, v16bf b, v8f c) {
  // D = A(16x32) * B(32x16) + C, bf16 in / f32 out
  return __builtin_amdgcn_wmma_f32_16x16x32_bf16(false, a, false, b, (short)0, c, false, false);
}

__device__ __forceinline__ float silu_f(float x) { return x / (1.0f + __expf(-x)); }

// ---------------------------------------------------------------------------
// LayerNorm: src is (l, b, D) f32; dst is token-major (n = b*L + l) f32.
// ---------------------------------------------------------------------------
__global__ void __launch_bounds__(256) ln_kernel(const float* __restrict__ src,
                                                 const float* __restrict__ w,
                                                 const float* __restrict__ bvec,
                                                 float* __restrict__ dst) {
  int n   = blockIdx.x;
  int tid = threadIdx.x;
  int l = n & (L_SEQ - 1);
  int b = n >> 11;
  const float* row = src + ((size_t)l * BATCH + b) * DMODEL;
  __shared__ float s1[256], s2[256];
  float a = 0.f, q = 0.f;
  for (int d = tid; d < DMODEL; d += 256) { float x = row[d]; a += x; q += x * x; }
  s1[tid] = a; s2[tid] = q;
  __syncthreads();
  for (int st = 128; st > 0; st >>= 1) {
    if (tid < st) { s1[tid] += s1[tid + st]; s2[tid] += s2[tid + st]; }
    __syncthreads();
  }
  float mu  = s1[0] / (float)DMODEL;
  float var = s2[0] / (float)DMODEL - mu * mu;
  float inv = rsqrtf(var + LN_EPS);
  for (int d = tid; d < DMODEL; d += 256)
    dst[(size_t)n * DMODEL + d] = (row[d] - mu) * inv * w[d] + bvec[d];
}

// ---------------------------------------------------------------------------
// Fragment layouts (flat storage: (tile*32 + lane) v16bf units)
//   A (MxK, 16x32 tiles):  elem j of lane -> row = lane&15,
//                          col = tk*32 + (j&7) + 8*(lane>>4) + 16*(j>>3)
//   B (KxN, 32x16 tiles):  elem j of lane -> k = tk*32 + (j&7) + 8*(lane>>4) + 16*(j>>3),
//                          col = tn*16 + (lane&15)
//   C (16x16 f32, 8 vgpr): vgpr r of lane -> row = r + 8*(lane>>4), col = lane&15
// ---------------------------------------------------------------------------

// h = silu(hprev*cw0 + hcur*cw1) fused into the A-fragment pack.
__global__ void __launch_bounds__(256) pack_hA_kernel(const float* __restrict__ hln,
                                                      const float* __restrict__ cw,
                                                      __bf16* __restrict__ dst) {
  const int TK = DMODEL >> 5;
  int t   = blockIdx.x * blockDim.x + threadIdx.x;
  int tot = (NTOK >> 4) * TK * 32;
  if (t >= tot) return;
  int lane = t & 31, tile = t >> 5;
  int tm = tile / TK, tk = tile % TK;
  int half = lane >> 4, li = lane & 15;
  int row = tm * 16 + li;            // token id
  int l   = row & (L_SEQ - 1);
  v16bf frag;
#pragma unroll
  for (int j = 0; j < 16; ++j) {
    int col  = tk * 32 + (j & 7) + 8 * half + 16 * (j >> 3);
    float hc = hln[(size_t)row * DMODEL + col];
    float hp = (l > 0) ? hln[(size_t)(row - 1) * DMODEL + col] : 0.f;
    frag[j]  = (__bf16)silu_f(hp * cw[col * 2] + hc * cw[col * 2 + 1]);
  }
  ((v16bf*)dst)[(size_t)tile * 32 + lane] = frag;
}

// Generic weight pack: f32 (K x M) row-major -> B fragments.
__global__ void __launch_bounds__(256) pack_B_kernel(const float* __restrict__ src,
                                                     __bf16* __restrict__ dst,
                                                     int K, int M) {
  int TN  = M >> 4;
  int tot = (K >> 5) * TN * 32;
  int t   = blockIdx.x * blockDim.x + threadIdx.x;
  if (t >= tot) return;
  int lane = t & 31, tile = t >> 5;
  int tk = tile / TN, tn = tile % TN;
  int half = lane >> 4, li = lane & 15;
  int col = tn * 16 + li;
  v16bf frag;
#pragma unroll
  for (int j = 0; j < 16; ++j) {
    int k   = tk * 32 + (j & 7) + 8 * half + 16 * (j >> 3);
    frag[j] = (__bf16)src[(size_t)k * M + col];
  }
  ((v16bf*)dst)[(size_t)tile * 32 + lane] = frag;
}

// A pack of f(s1) * (s2 ? s2 : 1) from token-major f32 (N x C).
template <int ACT1>
__global__ void __launch_bounds__(256) pack_A_mul_kernel(const float* __restrict__ s1,
                                                         const float* __restrict__ s2,
                                                         __bf16* __restrict__ dst,
                                                         int N, int C) {
  int TK  = C >> 5;
  int tot = (N >> 4) * TK * 32;
  int t   = blockIdx.x * blockDim.x + threadIdx.x;
  if (t >= tot) return;
  int lane = t & 31, tile = t >> 5;
  int tm = tile / TK, tk = tile % TK;
  int half = lane >> 4, li = lane & 15;
  int row = tm * 16 + li;
  v16bf frag;
#pragma unroll
  for (int j = 0; j < 16; ++j) {
    int col  = tk * 32 + (j & 7) + 8 * half + 16 * (j >> 3);
    size_t i = (size_t)row * C + col;
    float v  = s1[i];
    if (ACT1) v = silu_f(v);
    if (s2) v *= s2[i];
    frag[j] = (__bf16)v;
  }
  ((v16bf*)dst)[(size_t)tile * 32 + lane] = frag;
}

// ---------------------------------------------------------------------------
// WMMA GEMM with async global->LDS staging of B (CDNA5 ASYNCcnt pipeline).
// Block = 256 threads (8 waves) computes a 256x64 tile: wave w owns two
// 16-row tiles (register blocking), B fragments for the 64-col strip are
// staged once per K-step into LDS (double buffered) and reused by all waves
// and both row tiles -> 8 wmma per wave per K-step.
// ACT: 0=none, 1=silu, 2=residual-add with (l,b,D)-transposed store.
// Requires N % 256 == 0, M % 64 == 0 (true for all uses here).
// ---------------------------------------------------------------------------
template <int ACT>
__global__ void __launch_bounds__(256, 1)
gemm_kernel(const __bf16* __restrict__ A, const __bf16* __restrict__ B,
            float* __restrict__ out, const float* __restrict__ res,
            int N, int K, int M) {
  __shared__ __bf16 bsm[2 * 4 * 32 * 16];  // 2 buffers x 4 B-tiles x (32 lanes x 16 bf16) = 8KB
  const int TK = K >> 5, TN = M >> 4, TN4 = M >> 6;
  const int tid  = threadIdx.x;
  const int wave = tid >> 5, lane = tid & 31;
  const int blkRow = (int)blockIdx.x / TN4;   // 256-row block index
  const int c4     = (int)blockIdx.x % TN4;   // 64-col strip index
  const int rt0    = blkRow * 16 + wave * 2;  // this wave's two 16-row tiles
  const int rt1    = rt0 + 1;
  const v16bf* Af = (const v16bf*)A;

  // Each thread stages one 16-byte chunk per K-step: 4 tiles x 64 chunks.
  const int stTile  = tid >> 6;       // 0..3
  const int stChunk = tid & 63;       // chunk within tile
  const int stLane  = stChunk >> 1;   // fragment lane 0..31
  const int stHalf  = stChunk & 1;    // low/high 16B of the 32B lane record
  const unsigned ldsChunkOff = (unsigned)(((stTile * 32 + stLane) * 32) + stHalf * 16);
  const unsigned bufBytes    = 4u * 32u * 32u;  // 4KB per buffer (bsm assumed at LDS offset 0)

  auto stage = [&](int kt, int buf) {
    const char* g = (const char*)B +
        (((size_t)kt * TN + c4 * 4 + stTile) * 32 + (size_t)stLane) * 32 + stHalf * 16;
    unsigned ldsAddr = (unsigned)buf * bufBytes + ldsChunkOff;
    asm volatile("global_load_async_to_lds_b128 %0, %1, off"
                 :: "v"(ldsAddr), "v"((unsigned long long)(uintptr_t)g)
                 : "memory");
  };

  stage(0, 0);
  v8f acc[2][4] = {};
  for (int kt = 0; kt < TK; ++kt) {
    int cur = kt & 1;
    if (kt + 1 < TK) stage(kt + 1, cur ^ 1);
    v16bf a0 = Af[((size_t)rt0 * TK + kt) * 32 + lane];
    v16bf a1 = Af[((size_t)rt1 * TK + kt) * 32 + lane];
    if (kt + 1 < TK) {
      __builtin_prefetch(&Af[((size_t)rt0 * TK + kt + 1) * 32 + lane], 0, 3);
      __builtin_prefetch(&Af[((size_t)rt1 * TK + kt + 1) * 32 + lane], 0, 3);
    }
    if (kt + 1 < TK) asm volatile("s_wait_asynccnt 0x1" ::: "memory");
    else             asm volatile("s_wait_asynccnt 0x0" ::: "memory");
    __syncthreads();  // staged buffer visible to all 8 waves
    const v16bf* bt = (const v16bf*)(bsm + (size_t)cur * (4 * 32 * 16));
#pragma unroll
    for (int nn = 0; nn < 4; ++nn) {
      v16bf bb = bt[nn * 32 + lane];
      acc[0][nn] = wmma_bf16(a0, bb, acc[0][nn]);
      acc[1][nn] = wmma_bf16(a1, bb, acc[1][nn]);
    }
    __syncthreads();  // all reads done before this buffer is overwritten
  }

  int half = lane >> 4, li = lane & 15;
#pragma unroll
  for (int rr = 0; rr < 2; ++rr) {
    int rt = (rr == 0) ? rt0 : rt1;
#pragma unroll
    for (int nn = 0; nn < 4; ++nn) {
      int col = (c4 * 4 + nn) * 16 + li;
#pragma unroll
      for (int r = 0; r < 8; ++r) {
        int row = rt * 16 + r + 8 * half;
        float v = acc[rr][nn][r];
        if (ACT == 1) v = silu_f(v);
        if (ACT == 2) {
          int l = row & (L_SEQ - 1), b = row >> 11;
          size_t idx = ((size_t)l * BATCH + b) * M + col;
          out[idx] = v + res[idx];
        } else {
          out[(size_t)row * M + col] = v;
        }
      }
    }
  }
}

// ---------------------------------------------------------------------------
// RoPE on q and k, in place (token-major N x DK).
// ---------------------------------------------------------------------------
__global__ void __launch_bounds__(256) rope_kernel(float* __restrict__ q,
                                                   float* __restrict__ k) {
  int t   = blockIdx.x * blockDim.x + threadIdx.x;
  int tot = NTOK * HEADS * 16;
  if (t >= tot) return;
  int d = t & 15, rest = t >> 4;
  int h = rest & 15, n = rest >> 4;
  int l = n & (L_SEQ - 1);
  float inv = powf(10000.0f, -(float)d / 16.0f);
  float fr  = (float)l * inv;
  float c = cosf(fr), s = sinf(fr);
  size_t base = (size_t)n * DK_DIM + (size_t)h * HDKD;
  float a1 = q[base + d], a2 = q[base + 16 + d];
  q[base + d]      = a1 * c - a2 * s;
  q[base + 16 + d] = a2 * c + a1 * s;
  float b1 = k[base + d], b2 = k[base + 16 + d];
  k[base + d]      = b1 * c - b2 * s;
  k[base + 16 + d] = b2 * c + b1 * s;
}

// Router: logits->softmax->argmax->masked scores / onehot.
__global__ void __launch_bounds__(256) router_kernel(const float* __restrict__ v,
                                                     const float* __restrict__ rw,
                                                     float* __restrict__ ms,
                                                     float* __restrict__ mi) {
  int t   = blockIdx.x * blockDim.x + threadIdx.x;
  int tot = NTOK * HEADS;
  if (t >= tot) return;
  int h = t & 15, n = t >> 4;
  const float* vp = v + (size_t)n * DV_DIM + (size_t)h * HDVD;
  const float* wp = rw + (size_t)h * HDVD * 2;
  float l0 = 0.f, l1 = 0.f;
  for (int d = 0; d < HDVD; ++d) { float x = vp[d]; l0 += x * wp[d * 2]; l1 += x * wp[d * 2 + 1]; }
  float m  = fmaxf(l0, l1);
  float e0 = __expf(l0 - m), e1 = __expf(l1 - m);
  float den = e0 + e1;
  float p0 = e0 / den, p1 = e1 / den;
  int idx = (l1 > l0) ? 1 : 0;  // argmax, first wins on tie
  size_t o = ((size_t)n * HEADS + h) * 2;
  ms[o]     = (idx == 0) ? p0 : 0.f;
  ms[o + 1] = (idx == 1) ? p1 : 0.f;
  mi[o]     = (idx == 0) ? 1.f : 0.f;
  mi[o + 1] = (idx == 1) ? 1.f : 0.f;
}

// Inclusive cumsum over l per (b,h,r); qs = ms * (sum>0 ? 1/sum : 0).
__global__ void __launch_bounds__(32) scan_kernel(const float* __restrict__ ms,
                                                  const float* __restrict__ mi,
                                                  float* __restrict__ qs) {
  int bh = blockIdx.x;
  int r  = threadIdx.x;
  if (r >= 2) return;
  int b = bh >> 4, h = bh & 15;
  float sum = 0.f;
  for (int l = 0; l < L_SEQ; ++l) {
    size_t i = ((size_t)(b * L_SEQ + l) * HEADS + h) * 2 + r;
    sum += mi[i];
    qs[i] = ms[i] * (sum > 0.f ? 1.f / sum : 0.f);
  }
}

// Pack q_exp as A-frags per (b,h): L x 64  (col = r*32 + dk), scaled by qs.
__global__ void __launch_bounds__(256) pack_qexp_kernel(const float* __restrict__ q,
                                                        const float* __restrict__ qs,
                                                        __bf16* __restrict__ dst) {
  int t   = blockIdx.x * blockDim.x + threadIdx.x;
  int tot = 32 * 256 * 32;  // BH * tiles * lanes
  if (t >= tot) return;
  int lane = t & 31, rest = t >> 5;
  int bh = rest >> 8, tile = rest & 255;
  int tm = tile >> 1, tk = tile & 1;
  int b = bh >> 4, h = bh & 15;
  int half = lane >> 4, li = lane & 15;
  int lrow = tm * 16 + li;
  int n    = b * L_SEQ + lrow;
  v16bf frag;
#pragma unroll
  for (int j = 0; j < 16; ++j) {
    int col = tk * 32 + (j & 7) + 8 * half + 16 * (j >> 3);  // 0..63
    int r = col >> 5, dk = col & 31;
    float v = q[(size_t)n * DK_DIM + h * HDKD + dk] * qs[((size_t)n * HEADS + h) * 2 + r];
    frag[j] = (__bf16)v;
  }
  ((v16bf*)dst)[(size_t)bh * 8192 + (size_t)tile * 32 + lane] = frag;
}

// Pack k_exp^T as B-frags per (b,h): 64(d) x L(m), scaled by onehot mi.
__global__ void __launch_bounds__(256) pack_kexpT_kernel(const float* __restrict__ k,
                                                         const float* __restrict__ mi,
                                                         __bf16* __restrict__ dst) {
  int t   = blockIdx.x * blockDim.x + threadIdx.x;
  int tot = 32 * 256 * 32;
  if (t >= tot) return;
  int lane = t & 31, rest = t >> 5;
  int bh = rest >> 8, tile = rest & 255;   // tile = tk*128 + tn
  int tk = tile >> 7, tn = tile & 127;
  int b = bh >> 4, h = bh & 15;
  int half = lane >> 4, li = lane & 15;
  int m = tn * 16 + li;
  int n = b * L_SEQ + m;
  v16bf frag;
#pragma unroll
  for (int j = 0; j < 16; ++j) {
    int d = tk * 32 + (j & 7) + 8 * half + 16 * (j >> 3);  // 0..63
    int r = d >> 5, dk = d & 31;
    float v = k[(size_t)n * DK_DIM + h * HDKD + dk] * mi[((size_t)n * HEADS + h) * 2 + r];
    frag[j] = (__bf16)v;
  }
  ((v16bf*)dst)[(size_t)bh * 8192 + (size_t)tile * 32 + lane] = frag;
}

// Pack V as B-frags per (b,h): L(m) x 64(d).
__global__ void __launch_bounds__(256) pack_vB_kernel(const float* __restrict__ v,
                                                      __bf16* __restrict__ dst) {
  int t   = blockIdx.x * blockDim.x + threadIdx.x;
  int tot = 32 * 256 * 32;
  if (t >= tot) return;
  int lane = t & 31, rest = t >> 5;
  int bh = rest >> 8, tile = rest & 255;   // tile = tk*4 + tn
  int tk = tile >> 2, tn = tile & 3;
  int b = bh >> 4, h = bh & 15;
  int half = lane >> 4, li = lane & 15;
  int d = tn * 16 + li;
  v16bf frag;
#pragma unroll
  for (int j = 0; j < 16; ++j) {
    int m = tk * 32 + (j & 7) + 8 * half + 16 * (j >> 3);
    int n = b * L_SEQ + m;
    frag[j] = (__bf16)v[(size_t)n * DV_DIM + h * HDVD + d];
  }
  ((v16bf*)dst)[(size_t)bh * 8192 + (size_t)tile * 32 + lane] = frag;
}

// ---------------------------------------------------------------------------
// Attention: per (b,h) wave computes a 32-row strip (two 16-row tiles) of
// O = (S.*causal) @ V with S = Qexp @ KexpT * HDK^-0.5.  K/V fragments are
// reused across both row tiles; fused per-head LayerNorm at the end
// (32 lanes -> 32 rows).  launch_bounds(32,1): keep all fragments in VGPRs.
// ---------------------------------------------------------------------------
__global__ void __launch_bounds__(32, 1)
attn_kernel(const __bf16* __restrict__ qF, const __bf16* __restrict__ kF,
            const __bf16* __restrict__ vF, float* __restrict__ onorm) {
  __shared__ float smem[2 * 16 * 64];  // per-tile S staging (16x32) + O staging (2x16x64)
  int bh = blockIdx.x;
  int rtBase = blockIdx.y * 2;         // two 16-row tiles: rtBase, rtBase+1
  int lane = threadIdx.x;
  int half = lane >> 4, li = lane & 15;
  const v16bf* qf = (const v16bf*)(qF + (size_t)bh * L_SEQ * 64);
  const v16bf* kf = (const v16bf*)(kF + (size_t)bh * L_SEQ * 64);
  const v16bf* vf = (const v16bf*)(vF + (size_t)bh * L_SEQ * 64);
  v16bf qa[2][2];
#pragma unroll
  for (int t2 = 0; t2 < 2; ++t2)
#pragma unroll
    for (int kt = 0; kt < 2; ++kt)
      qa[t2][kt] = qf[(size_t)((rtBase + t2) * 2 + kt) * 32 + lane];
  v8f oc[2][4] = {};
  const float scale = 0.17677669529663687f;  // 1/sqrt(32)
  int nchunks = (int)blockIdx.y + 1;         // chunks of 32 up to row rtBase*16+31
  for (int mc = 0; mc < nchunks; ++mc) {
    if (mc + 1 < nchunks) {  // global_prefetch_b8 next K/V chunk
      __builtin_prefetch(&kf[(size_t)(mc * 2 + 2) * 32 + lane], 0, 3);
      __builtin_prefetch(&kf[(size_t)(128 + mc * 2 + 2) * 32 + lane], 0, 3);
      __builtin_prefetch(&vf[(size_t)(mc * 4 + 4) * 32 + lane], 0, 3);
    }
#pragma unroll
    for (int t2 = 0; t2 < 2; ++t2) {
#pragma unroll
      for (int tt = 0; tt < 2; ++tt) {
        v8f s = {};
        s = wmma_bf16(qa[t2][0], kf[(size_t)(0 * 128 + mc * 2 + tt) * 32 + lane], s);
        s = wmma_bf16(qa[t2][1], kf[(size_t)(1 * 128 + mc * 2 + tt) * 32 + lane], s);
        int m = mc * 32 + tt * 16 + li;
        int rowbase = (rtBase + t2) * 16 + 8 * half;
#pragma unroll
        for (int r = 0; r < 8; ++r) {
          float val = (m <= rowbase + r) ? s[r] * scale : 0.0f;  // causal multiply-mask
          smem[t2 * 512 + (r + 8 * half) * 32 + tt * 16 + li] = val;
        }
      }
    }
    __syncthreads();
#pragma unroll
    for (int t2 = 0; t2 < 2; ++t2) {
      v16bf p;  // re-layout S(16x32) into an A fragment for P@V
#pragma unroll
      for (int j = 0; j < 16; ++j) {
        int col = (j & 7) + 8 * half + 16 * (j >> 3);
        p[j] = (__bf16)smem[t2 * 512 + li * 32 + col];
      }
#pragma unroll
      for (int nn = 0; nn < 4; ++nn)
        oc[t2][nn] = wmma_bf16(p, vf[(size_t)(mc * 4 + nn) * 32 + lane], oc[t2][nn]);
    }
    __syncthreads();
  }
  // stage O(2 x 16x64) and do per-row (per-head) LayerNorm; 32 lanes = 32 rows
#pragma unroll
  for (int t2 = 0; t2 < 2; ++t2)
#pragma unroll
    for (int nn = 0; nn < 4; ++nn)
#pragma unroll
      for (int r = 0; r < 8; ++r)
        smem[t2 * 1024 + (r + 8 * half) * 64 + nn * 16 + li] = oc[t2][nn][r];
  __syncthreads();
  {
    int t2 = lane >> 4, rloc = lane & 15;
    const float* rowp = &smem[t2 * 1024 + rloc * 64];
    float mu = 0.f, m2 = 0.f;
    for (int d = 0; d < 64; ++d) { float x = rowp[d]; mu += x; m2 += x * x; }
    mu *= (1.f / 64.f);
    float var = m2 * (1.f / 64.f) - mu * mu;
    float inv = rsqrtf(var + LN_EPS);
    int b = bh >> 4, h = bh & 15;
    int lrow = (rtBase + t2) * 16 + rloc;
    size_t base = ((size_t)(b * L_SEQ + lrow)) * DV_DIM + (size_t)h * HDVD;
    for (int d = 0; d < 64; ++d)
      onorm[base + d] = (rowp[d] - mu) * inv;
  }
}

// ---------------------------------------------------------------------------
// Host-side orchestration
// ---------------------------------------------------------------------------
extern "C" void kernel_launch(void* const* d_in, const int* in_sizes, int n_in,
                              void* d_out, int out_size, void* d_ws, size_t ws_size,
                              hipStream_t stream) {
  (void)in_sizes; (void)n_in; (void)out_size; (void)ws_size;
  const float* x      = (const float*)d_in[0];
  const float* ln1_w  = (const float*)d_in[2];
  const float* ln1_b  = (const float*)d_in[3];
  const float* ln2_w  = (const float*)d_in[4];
  const float* ln2_b  = (const float*)d_in[5];
  const float* conv_w = (const float*)d_in[6];
  const float* Wq     = (const float*)d_in[7];
  const float* Wk     = (const float*)d_in[8];
  const float* Wv     = (const float*)d_in[9];
  const float* Wg     = (const float*)d_in[10];
  const float* Wout   = (const float*)d_in[11];
  const float* rw     = (const float*)d_in[12];
  const float* w1     = (const float*)d_in[13];
  const float* w3     = (const float*)d_in[14];
  const float* w2     = (const float*)d_in[15];

  char* ws = (char*)d_ws;
  size_t off = 0;
  auto alloc = [&](size_t bytes) -> void* {
    void* p = ws + off;
    off += (bytes + 255) & ~(size_t)255;
    return p;
  };
  float*  hln    = (float*)alloc((size_t)NTOK * DMODEL * 4);   // also reused for h2
  __bf16* hA     = (__bf16*)alloc((size_t)NTOK * DMODEL * 2);
  __bf16* wqF    = (__bf16*)alloc((size_t)DMODEL * DK_DIM * 2);
  __bf16* wkF    = (__bf16*)alloc((size_t)DMODEL * DK_DIM * 2);
  __bf16* wvF    = (__bf16*)alloc((size_t)DMODEL * DV_DIM * 2);
  __bf16* wgF    = (__bf16*)alloc((size_t)DMODEL * DV_DIM * 2);
  __bf16* woutF  = (__bf16*)alloc((size_t)DV_DIM * DMODEL * 2);
  __bf16* w1F    = (__bf16*)alloc((size_t)DMODEL * FF_DIM * 2);
  __bf16* w3F    = (__bf16*)alloc((size_t)DMODEL * FF_DIM * 2);
  __bf16* w2F    = (__bf16*)alloc((size_t)FF_DIM * DMODEL * 2);
  float*  qbuf   = (float*)alloc((size_t)NTOK * DK_DIM * 4);
  float*  kbuf   = (float*)alloc((size_t)NTOK * DK_DIM * 4);
  float*  vbuf   = (float*)alloc((size_t)NTOK * DV_DIM * 4);
  float*  gbuf   = (float*)alloc((size_t)NTOK * DV_DIM * 4);
  float*  ms     = (float*)alloc((size_t)NTOK * HEADS * 2 * 4);
  float*  mi     = (float*)alloc((size_t)NTOK * HEADS * 2 * 4);
  float*  qs     = (float*)alloc((size_t)NTOK * HEADS * 2 * 4);
  __bf16* qexpF  = (__bf16*)alloc((size_t)32 * L_SEQ * 64 * 2);
  __bf16* kexpF  = (__bf16*)alloc((size_t)32 * L_SEQ * 64 * 2);
  __bf16* vFr    = (__bf16*)alloc((size_t)32 * L_SEQ * 64 * 2);
  float*  onorm  = (float*)alloc((size_t)NTOK * DV_DIM * 4);
  __bf16* gatedA = (__bf16*)alloc((size_t)NTOK * DV_DIM * 2);
  float*  attnO  = (float*)alloc((size_t)NTOK * DMODEL * 4);   // (l,b,D) attn_out
  __bf16* h2A    = (__bf16*)alloc((size_t)NTOK * DMODEL * 2);
  float*  ff1    = (float*)alloc((size_t)NTOK * FF_DIM * 4);
  float*  ff3    = (float*)alloc((size_t)NTOK * FF_DIM * 4);
  __bf16* ffA    = (__bf16*)alloc((size_t)NTOK * FF_DIM * 2);

  auto cdiv = [](long long a, long long b) { return (int)((a + b - 1) / b); };
  // async GEMM: one block per (256-row x 64-col) tile, exact cover
  auto gemm_blocks = [&](int N, int M) { return (N / 256) * (M / 64); };

  // 1. LN1 (with (l,b,D) -> token-major transpose)
  ln_kernel<<<NTOK, 256, 0, stream>>>(x, ln1_w, ln1_b, hln);

  // 2. conv + silu fused into A-fragment pack of h
  pack_hA_kernel<<<cdiv((long long)(NTOK / 16) * (DMODEL / 32) * 32, 256), 256, 0, stream>>>(hln, conv_w, hA);

  // 3. weight packs (bf16 B-fragments)
  auto packB = [&](const float* s, __bf16* d, int K, int M) {
    pack_B_kernel<<<cdiv((long long)(K / 32) * (M / 16) * 32, 256), 256, 0, stream>>>(s, d, K, M);
  };
  packB(Wq, wqF, DMODEL, DK_DIM);
  packB(Wk, wkF, DMODEL, DK_DIM);
  packB(Wv, wvF, DMODEL, DV_DIM);
  packB(Wg, wgF, DMODEL, DV_DIM);
  packB(Wout, woutF, DV_DIM, DMODEL);
  packB(w1, w1F, DMODEL, FF_DIM);
  packB(w3, w3F, DMODEL, FF_DIM);
  packB(w2, w2F, FF_DIM, DMODEL);

  // 4. projections (q gets fused SiLU)
  gemm_kernel<1><<<gemm_blocks(NTOK, DK_DIM), 256, 0, stream>>>(hA, wqF, qbuf, nullptr, NTOK, DMODEL, DK_DIM);
  gemm_kernel<0><<<gemm_blocks(NTOK, DK_DIM), 256, 0, stream>>>(hA, wkF, kbuf, nullptr, NTOK, DMODEL, DK_DIM);
  gemm_kernel<0><<<gemm_blocks(NTOK, DV_DIM), 256, 0, stream>>>(hA, wvF, vbuf, nullptr, NTOK, DMODEL, DV_DIM);
  gemm_kernel<0><<<gemm_blocks(NTOK, DV_DIM), 256, 0, stream>>>(hA, wgF, gbuf, nullptr, NTOK, DMODEL, DV_DIM);

  // 5. RoPE on q, k
  rope_kernel<<<cdiv((long long)NTOK * HEADS * 16, 256), 256, 0, stream>>>(qbuf, kbuf);

  // 6-7. router + cumsum scan
  router_kernel<<<cdiv((long long)NTOK * HEADS, 256), 256, 0, stream>>>(vbuf, rw, ms, mi);
  scan_kernel<<<32, 32, 0, stream>>>(ms, mi, qs);

  // 8. expansion packs (q_exp A, k_exp^T B, V B) per (b,h)
  pack_qexp_kernel<<<cdiv(32LL * 256 * 32, 256), 256, 0, stream>>>(qbuf, qs, qexpF);
  pack_kexpT_kernel<<<cdiv(32LL * 256 * 32, 256), 256, 0, stream>>>(kbuf, mi, kexpF);
  pack_vB_kernel<<<cdiv(32LL * 256 * 32, 256), 256, 0, stream>>>(vbuf, vFr);

  // 9. attention (causal masked-mul, no softmax) + fused per-head LN
  attn_kernel<<<dim3(32, L_SEQ / 32), 32, 0, stream>>>(qexpF, kexpF, vFr, onorm);

  // 10. gating: silu(g) * o_norm -> A-fragments
  pack_A_mul_kernel<1><<<cdiv((long long)(NTOK / 16) * (DV_DIM / 32) * 32, 256), 256, 0, stream>>>(
      gbuf, onorm, gatedA, NTOK, DV_DIM);

  // 11. output projection + residual (stores (l,b,D))
  gemm_kernel<2><<<gemm_blocks(NTOK, DMODEL), 256, 0, stream>>>(gatedA, woutF, attnO, x, NTOK, DV_DIM, DMODEL);

  // 12-13. LN2 + pack
  ln_kernel<<<NTOK, 256, 0, stream>>>(attnO, ln2_w, ln2_b, hln);
  pack_A_mul_kernel<0><<<cdiv((long long)(NTOK / 16) * (DMODEL / 32) * 32, 256), 256, 0, stream>>>(
      hln, nullptr, h2A, NTOK, DMODEL);

  // 14. SwiGLU MLP
  gemm_kernel<1><<<gemm_blocks(NTOK, FF_DIM), 256, 0, stream>>>(h2A, w1F, ff1, nullptr, NTOK, DMODEL, FF_DIM);
  gemm_kernel<0><<<gemm_blocks(NTOK, FF_DIM), 256, 0, stream>>>(h2A, w3F, ff3, nullptr, NTOK, DMODEL, FF_DIM);
  pack_A_mul_kernel<0><<<cdiv((long long)(NTOK / 16) * (FF_DIM / 32) * 32, 256), 256, 0, stream>>>(
      ff1, ff3, ffA, NTOK, FF_DIM);

  // 15. down-projection + residual -> final output (l,b,D) f32
  gemm_kernel<2><<<gemm_blocks(NTOK, DMODEL), 256, 0, stream>>>(ffA, w2F, (float*)d_out, attnO, NTOK, FF_DIM, DMODEL);
}
